// BaseProcessingLSTM_52699248722559
// MI455X (gfx1250) — matrix operand order repaired and checked by reading, not compiled, not run on hardware
//
#include <hip/hip_runtime.h>
#include <hip/hip_bf16.h>

// ---------------------------------------------------------------------------
// 2-layer LSTM on MI455X (gfx1250), bf16 WMMA path.
//  B=32, T=512, IN=128, H=1024, L=2, OUT=128
//
//  * Pre-pass: convert weights AND x to bf16 (native v_cvt, no hot-path
//    conversions); precompute embed GEMM for all T (no recurrence).
//  * Per step/layer: 64 WGs x 256 thr. WG owns 16 hidden columns.
//    wave = (gate q, K-half kh); each wave computes BOTH m-tiles so every
//    weight fragment is loaded exactly once per WG (B-traffic optimal).
//    K-half partials reduced through LDS, fused LSTM cell in-WG.
//  * h ping-pongs (bf16), c in place (f32, column-owned).
// ---------------------------------------------------------------------------

#define BZ  32
#define TT  512
#define IND 128
#define HD  1024
#define LL  2
#define OUTD 128

typedef __attribute__((ext_vector_type(16))) __bf16 v16bf;
typedef __attribute__((ext_vector_type(8)))  __bf16 v8bf;
typedef __attribute__((ext_vector_type(8)))  float  v8f;
typedef __attribute__((ext_vector_type(4)))  float  v4f;

union Frag16 {
    v16bf v;
    v8bf  h[2];
};

__device__ __forceinline__ float sigm(float x) {
    return 1.0f / (1.0f + __expf(-x));
}
__device__ __forceinline__ float tanh_fast(float x) {
    float e = __expf(2.0f * x);
    return 1.0f - 2.0f / (e + 1.0f);
}

// A fragment: 16x32 bf16, M = lane&15, K-halves selected by lane>>4.
__device__ __forceinline__ v16bf load_a(const __bf16* row_base, int k0, int lane) {
    int koff = k0 + ((lane >> 4) << 4);
    Frag16 f;
    f.h[0] = *reinterpret_cast<const v8bf*>(row_base + koff);
    f.h[1] = *reinterpret_cast<const v8bf*>(row_base + koff + 8);
    return f.v;
}

// B fragment: 32x16 bf16 (KxN). Weights row-major [rows][K]: the lane's 16
// consecutive K values are one contiguous 32-byte load.
__device__ __forceinline__ v16bf load_b(const __bf16* w, int row, int K, int k0, int lane) {
    const __bf16* p = w + (size_t)row * K + k0 + ((lane >> 4) << 4);
    return *reinterpret_cast<const v16bf*>(p);
}

#define WMMA_BF16(A, Bm, C) \
    __builtin_amdgcn_wmma_f32_16x16x32_bf16(false, (A), false, (Bm), (short)0, (C), false, false)

// ------------------------------------------------------- f32 -> bf16 (x8)
__global__ void cvt_f32_bf16_x8(const float* __restrict__ src,
                                __bf16* __restrict__ dst, int n8) {
    int i = blockIdx.x * blockDim.x + threadIdx.x;
    int stride = gridDim.x * blockDim.x;
    for (; i < n8; i += stride) {
        const v4f* s = reinterpret_cast<const v4f*>(src + (size_t)i * 8);
        v4f a = s[0], b = s[1];
        v8bf o;
        o[0] = (__bf16)a[0]; o[1] = (__bf16)a[1];
        o[2] = (__bf16)a[2]; o[3] = (__bf16)a[3];
        o[4] = (__bf16)b[0]; o[5] = (__bf16)b[1];
        o[6] = (__bf16)b[2]; o[7] = (__bf16)b[3];
        *reinterpret_cast<v8bf*>(dst + (size_t)i * 8) = o;
    }
}

// h0 layout: [B, L*2*H] = per layer (h, c). h -> bf16 ping0, c -> f32.
__global__ void init_state(const float* __restrict__ h0,
                           __bf16* __restrict__ hbuf,   // [L][2][B][H]
                           float* __restrict__ cbuf) {  // [L][B][H]
    int i = blockIdx.x * blockDim.x + threadIdx.x;
    int stride = gridDim.x * blockDim.x;
    int n = LL * BZ * HD;
    for (; i < n; i += stride) {
        int l = i / (BZ * HD);
        int r = i - l * (BZ * HD);
        int b = r / HD;
        int k = r - b * HD;
        const float* src = h0 + (size_t)b * (LL * 2 * HD) + l * (2 * HD);
        hbuf[((size_t)(l * 2 + 0) * BZ + b) * HD + k] = (__bf16)src[k];
        cbuf[((size_t)l * BZ + b) * HD + k] = src[HD + k];
    }
}

// -------------------------------------------------------------- embed GEMM
// emb[t][b][n] = bf16( x[b][t][:] @ Wemb[n][:] + b_embed[n] )
// grid = T*8 WGs of 256; wave -> one n-tile, BOTH m-tiles (B frag reused).
__global__ void embed_gemm(const __bf16* __restrict__ xbf,  // [B][T][IN] bf16
                           const __bf16* __restrict__ wemb, // [H][IN] bf16
                           const float* __restrict__ bemb,
                           __bf16* __restrict__ emb) {      // [T][B][H]
    int t    = blockIdx.x >> 3;
    int nblk = blockIdx.x & 7;
    int wave = threadIdx.x >> 5;
    int lane = threadIdx.x & 31;
    int n0 = (nblk * 8 + wave) * 16;

    int brow  = lane & 15;
    int khalf = lane >> 4;
    const __bf16* a0_base = xbf + ((size_t)brow * TT + t) * IND;
    const __bf16* a1_base = xbf + ((size_t)(16 + brow) * TT + t) * IND;

    v8f acc0 = {}, acc1 = {};
    #pragma unroll
    for (int kk = 0; kk < IND / 32; ++kk) {
        int k0 = kk * 32;
        v16bf bf = load_b(wemb, n0 + brow, IND, k0, lane);
        v16bf a0 = load_a(a0_base, k0, lane);
        v16bf a1 = load_a(a1_base, k0, lane);
        acc0 = WMMA_BF16(a0, bf, acc0);
        acc1 = WMMA_BF16(a1, bf, acc1);
    }

    int   col  = n0 + brow;
    float bias = bemb[col];
    #pragma unroll
    for (int r = 0; r < 8; ++r) {
        int row = r + (khalf << 3);
        emb[((size_t)t * BZ + row) * HD + col]        = (__bf16)(acc0[r] + bias);
        emb[((size_t)t * BZ + (16 + row)) * HD + col] = (__bf16)(acc1[r] + bias);
    }
}

// ----------------------------------------------------------- LSTM gate step
// gates = h_in@Wih^T + h_prev@Whh^T + bih + bhh, fused pointwise cell.
// grid = 64 WGs x 256 threads. WG owns 16 hidden columns (col0 = blk*16).
// wave w: q = w&3 (gate i/f/g/o), kh = w>>2 (K-half). Each wave: both m-tiles.
__global__ void lstm_gates(const __bf16* __restrict__ h_in,   // [B][H]
                           const __bf16* __restrict__ h_prev, // [B][H]
                           __bf16* __restrict__ h_out,        // [B][H]
                           float* __restrict__ c,             // [B][H]
                           const __bf16* __restrict__ wih,    // [4H][H]
                           const __bf16* __restrict__ whh,    // [4H][H]
                           const float* __restrict__ bih,
                           const float* __restrict__ bhh) {
    __shared__ float gsm[2][4][BZ][16];   // [K-half][gate][row][col]

    int col0 = blockIdx.x * 16;
    int wave = threadIdx.x >> 5;
    int lane = threadIdx.x & 31;
    int q  = wave & 3;
    int kh = wave >> 2;
    int brow = lane & 15;

    int wrow = q * HD + col0 + brow;
    const __bf16* wih_row = wih + (size_t)wrow * HD;
    const __bf16* whh_row = whh + (size_t)wrow * HD;
    const __bf16* ain0 = h_in   + (size_t)brow * HD;
    const __bf16* ain1 = h_in   + (size_t)(16 + brow) * HD;
    const __bf16* ah0  = h_prev + (size_t)brow * HD;
    const __bf16* ah1  = h_prev + (size_t)(16 + brow) * HD;

    v8f acc0 = {}, acc1 = {};
    int kbase = kh * (HD / 2);
    #pragma unroll 4
    for (int kk = 0; kk < HD / 64; ++kk) {
        int k0   = kbase + kk * 32;
        int koffB = k0 + ((lane >> 4) << 4);
        // Prefetch next iteration's weight rows into L0 (streamed from L2).
        __builtin_prefetch(wih_row + koffB + 32, 0, 3);
        __builtin_prefetch(whh_row + koffB + 32, 0, 3);

        v16bf b_ih = *reinterpret_cast<const v16bf*>(wih_row + koffB);
        v16bf a_in0 = load_a(ain0, k0, lane);
        v16bf a_in1 = load_a(ain1, k0, lane);
        acc0 = WMMA_BF16(a_in0, b_ih, acc0);
        acc1 = WMMA_BF16(a_in1, b_ih, acc1);

        v16bf b_hh = *reinterpret_cast<const v16bf*>(whh_row + koffB);
        v16bf a_h0 = load_a(ah0, k0, lane);
        v16bf a_h1 = load_a(ah1, k0, lane);
        acc0 = WMMA_BF16(a_h0, b_hh, acc0);
        acc1 = WMMA_BF16(a_h1, b_hh, acc1);
    }

    int khalf = lane >> 4;
    #pragma unroll
    for (int r = 0; r < 8; ++r) {
        int row = r + (khalf << 3);
        gsm[kh][q][row][brow]      = acc0[r];
        gsm[kh][q][16 + row][brow] = acc1[r];
    }
    __syncthreads();

    // Fused LSTM cell: 512 elements (32 rows x 16 cols), 2 per thread.
    for (int e = threadIdx.x; e < BZ * 16; e += blockDim.x) {
        int row  = e >> 4;
        int cc   = e & 15;
        int colg = col0 + cc;
        float gi = gsm[0][0][row][cc] + gsm[1][0][row][cc] + bih[0 * HD + colg] + bhh[0 * HD + colg];
        float gf = gsm[0][1][row][cc] + gsm[1][1][row][cc] + bih[1 * HD + colg] + bhh[1 * HD + colg];
        float gg = gsm[0][2][row][cc] + gsm[1][2][row][cc] + bih[2 * HD + colg] + bhh[2 * HD + colg];
        float go = gsm[0][3][row][cc] + gsm[1][3][row][cc] + bih[3 * HD + colg] + bhh[3 * HD + colg];
        float ig = sigm(gi);
        float fg = sigm(gf);
        float gt = tanh_fast(gg);
        float og = sigm(go);
        size_t idx = (size_t)row * HD + colg;
        float cn = fg * c[idx] + ig * gt;
        c[idx] = cn;
        h_out[idx] = (__bf16)(og * tanh_fast(cn));
    }
}

// ---------------------------------------------------------- output project
// out[b][t][:] = h1 @ Wout^T + b_out. grid = 1 WG x 8 waves (n-tiles),
// each wave does both m-tiles (B frag reused).
__global__ void out_proj(const __bf16* __restrict__ h,     // [B][H]
                         const __bf16* __restrict__ wout,  // [OUT][H]
                         const float* __restrict__ bout,
                         float* __restrict__ out, int t) {
    int wave = threadIdx.x >> 5;
    int lane = threadIdx.x & 31;
    int n0 = wave * 16;
    int brow = lane & 15;

    const __bf16* a0_base = h + (size_t)brow * HD;
    const __bf16* a1_base = h + (size_t)(16 + brow) * HD;
    const __bf16* w_row   = wout + (size_t)(n0 + brow) * HD;

    v8f acc0 = {}, acc1 = {};
    #pragma unroll 4
    for (int kk = 0; kk < HD / 32; ++kk) {
        int k0 = kk * 32;
        v16bf b  = *reinterpret_cast<const v16bf*>(w_row + k0 + ((lane >> 4) << 4));
        v16bf a0 = load_a(a0_base, k0, lane);
        v16bf a1 = load_a(a1_base, k0, lane);
        acc0 = WMMA_BF16(a0, b, acc0);
        acc1 = WMMA_BF16(a1, b, acc1);
    }

    int   col  = n0 + brow;
    float bias = bout[col];
    int khalf = lane >> 4;
    #pragma unroll
    for (int r = 0; r < 8; ++r) {
        int row = r + (khalf << 3);
        out[((size_t)row * TT + t) * OUTD + col]        = acc0[r] + bias;
        out[((size_t)(16 + row) * TT + t) * OUTD + col] = acc1[r] + bias;
    }
}

// ---------------------------------------------------------------------------
extern "C" void kernel_launch(void* const* d_in, const int* in_sizes, int n_in,
                              void* d_out, int out_size, void* d_ws, size_t ws_size,
                              hipStream_t stream) {
    const float* x       = (const float*)d_in[0];
    const float* W_embed = (const float*)d_in[1];
    const float* b_embed = (const float*)d_in[2];
    const float* W_ih    = (const float*)d_in[3];
    const float* b_ih    = (const float*)d_in[4];
    const float* W_hh    = (const float*)d_in[5];
    const float* b_hh    = (const float*)d_in[6];
    const float* W_out   = (const float*)d_in[7];
    const float* b_out   = (const float*)d_in[8];
    const float* h0      = (const float*)d_in[9];
    float* out = (float*)d_out;

    // Workspace layout (69 MiB total).
    char* ws = (char*)d_ws;
    const size_t N_WEMB = (size_t)HD * IND;            // 131072
    const size_t N_WIH  = (size_t)LL * 4 * HD * HD;    // 8388608
    const size_t N_WOUT = (size_t)OUTD * HD;           // 131072
    const size_t N_X    = (size_t)BZ * TT * IND;       // 2097152
    const size_t N_EMB  = (size_t)TT * BZ * HD;        // 16777216
    const size_t N_HBUF = (size_t)LL * 2 * BZ * HD;    // 131072
    const size_t N_C    = (size_t)LL * BZ * HD;        // 65536

    __bf16* wemb = (__bf16*)ws;   ws += N_WEMB * 2;
    __bf16* wih  = (__bf16*)ws;   ws += N_WIH  * 2;
    __bf16* whh  = (__bf16*)ws;   ws += N_WIH  * 2;
    __bf16* wout = (__bf16*)ws;   ws += N_WOUT * 2;
    __bf16* xbf  = (__bf16*)ws;   ws += N_X    * 2;
    __bf16* emb  = (__bf16*)ws;   ws += N_EMB  * 2;
    __bf16* hbuf = (__bf16*)ws;   ws += N_HBUF * 2;
    float*  cbuf = (float*)ws;    ws += N_C    * 4;

    // 1) bf16 conversions (weights L2-resident afterwards: ~33.5 MB).
    cvt_f32_bf16_x8<<<64,   256, 0, stream>>>(W_embed, wemb, (int)(N_WEMB / 8));
    cvt_f32_bf16_x8<<<2048, 256, 0, stream>>>(W_ih,    wih,  (int)(N_WIH  / 8));
    cvt_f32_bf16_x8<<<2048, 256, 0, stream>>>(W_hh,    whh,  (int)(N_WIH  / 8));
    cvt_f32_bf16_x8<<<64,   256, 0, stream>>>(W_out,   wout, (int)(N_WOUT / 8));
    cvt_f32_bf16_x8<<<1024, 256, 0, stream>>>(x,       xbf,  (int)(N_X    / 8));

    // 2) Initial (h, c) state.
    init_state<<<256, 256, 0, stream>>>(h0, hbuf, cbuf);

    // 3) Embed projection for ALL timesteps at once (no recurrence).
    embed_gemm<<<TT * 8, 256, 0, stream>>>(xbf, wemb, b_embed, emb);

    // 4) Sequential time loop; h ping-pongs, c updated in place.
    auto hptr = [&](int l, int p) {
        return hbuf + ((size_t)(l * 2 + p) * BZ) * HD;
    };
    for (int t = 0; t < TT; ++t) {
        int p = t & 1;
        lstm_gates<<<64, 256, 0, stream>>>(
            emb + (size_t)t * BZ * HD,
            hptr(0, p), hptr(0, p ^ 1),
            cbuf,
            wih, whh, b_ih, b_hh);
        lstm_gates<<<64, 256, 0, stream>>>(
            hptr(0, p ^ 1),
            hptr(1, p), hptr(1, p ^ 1),
            cbuf + (size_t)BZ * HD,
            wih + (size_t)4 * HD * HD, whh + (size_t)4 * HD * HD,
            b_ih + 4 * HD, b_hh + 4 * HD);
        out_proj<<<1, 256, 0, stream>>>(hptr(1, p ^ 1), wout, b_out, out, t);
    }
}
